// SelfAttention_34299608826361
// MI455X (gfx1250) — compile-verified, matrix operand order checked by reading, not run on hardware
//
#include <hip/hip_runtime.h>
#include <hip/hip_bf16.h>

typedef __attribute__((ext_vector_type(16))) _Float16 v16h;
typedef __attribute__((ext_vector_type(8)))  _Float16 v8h;
typedef __attribute__((ext_vector_type(8)))  float    v8f;

#define DEV __device__ __forceinline__

// Use the CDNA5 Tensor Data Mover for K/V tile staging if the builtin exists.
// (Flip to 0 if the toolchain's builtin arity mismatches.)
#if defined(__has_builtin)
#  if __has_builtin(__builtin_amdgcn_tensor_load_to_lds)
#    define USE_TDM 1
#  endif
#endif
#ifndef USE_TDM
#  define USE_TDM 0
#endif

// ---------------------------------------------------------------------------
// WMMA wrapper: D = A(16x32 f16) * B(32x16 f16) + C(16x16 f32)
// ---------------------------------------------------------------------------
DEV v8f wmma_f16(v16h a, v16h b, v8f c) {
    return __builtin_amdgcn_wmma_f32_16x16x32_f16(
        false, a, false, b, (short)0, c, false, false);
}

// ---------------------------------------------------------------------------
// Fragment loader. 16-bit A 16x32 layout (ISA 7.12.2):
//   lanes 0-15 : row = lane,    K = 0..7  and 16..23
//   lanes 16-31: row = lane-16, K = 8..15 and 24..31
// B mirrors with N in place of M. Works for global or LDS pointers.
// ---------------------------------------------------------------------------
DEV v16h load_frag_rowmajor(const _Float16* base, int ld) {
    const int lane = threadIdx.x & 31;
    const int r    = lane & 15;
    const int koff = (lane >> 4) << 3;   // 0 or 8
    const _Float16* p = base + (size_t)r * ld + koff;
    union { v16h v; v8h h[2]; } f;
    f.h[0] = *(const v8h*)(p);
    f.h[1] = *(const v8h*)(p + 16);
    return f.v;
}

#if USE_TDM
typedef unsigned uv4 __attribute__((ext_vector_type(4)));
typedef int      iv8 __attribute__((ext_vector_type(8)));
typedef int      iv4 __attribute__((ext_vector_type(4)));

DEV unsigned lds_off_of(const void* p) {   // generic LDS ptr -> 32-bit LDS byte addr
    return (unsigned)(unsigned long long)p;
}

// 2D TDM load: tile (tile_d1 rows x tile_d0_dw dwords) from a row-major
// tensor with row stride stride_dw dwords, into contiguous LDS at lds_byte.
// D# bitfields per CDNA5 ISA ch.8 (group0 128b, group1 256b; groups 2/3 zero).
DEV void tdm_load_2d(unsigned lds_byte, const void* gsrc,
                     unsigned tile_d0_dw, unsigned tile_d1,
                     unsigned stride_dw, unsigned tensor_d0, unsigned tensor_d1) {
    const unsigned long long ga = (unsigned long long)gsrc;
    uv4 g0;
    g0[0] = 1u;                                  // count=1, user descriptor
    g0[1] = lds_byte;                            // lds_addr
    g0[2] = (unsigned)ga;                        // global_addr[31:0]
    g0[3] = (unsigned)(ga >> 32) | (2u << 30);   // global_addr[56:32] | type=2
    iv8 g1;
    g1[0] = (int)(2u << 16);                                   // data_size=4B
    g1[1] = (int)((tensor_d0 & 0xFFFFu) << 16);                // tensor_dim0 lo
    g1[2] = (int)((tensor_d0 >> 16) | ((tensor_d1 & 0xFFFFu) << 16));
    g1[3] = (int)((tensor_d1 >> 16) | (tile_d0_dw << 16));     // tile_dim0
    g1[4] = (int)(tile_d1 & 0xFFFFu);                          // tile_dim1
    g1[5] = (int)stride_dw;                                    // dim0_stride lo
    g1[6] = 0;
    g1[7] = 0;
    iv4 z4 = {0, 0, 0, 0};
#if defined(__clang_major__) && (__clang_major__ >= 23)
    iv8 z8 = {0, 0, 0, 0, 0, 0, 0, 0};
    __builtin_amdgcn_tensor_load_to_lds(g0, g1, z4, z4, z8, 0);
#else
    __builtin_amdgcn_tensor_load_to_lds(g0, g1, z4, z4, 0);
#endif
}
#endif  // USE_TDM

// ---------------------------------------------------------------------------
// Prep kernels
// ---------------------------------------------------------------------------
__global__ void cvt_f32_to_f16_kernel(const float* __restrict__ in,
                                      _Float16* __restrict__ out, int n) {
    int i = blockIdx.x * blockDim.x + threadIdx.x;
    if (i < n) out[i] = (_Float16)in[i];
}

__global__ void transpose_cvt_kernel(const float* __restrict__ W,
                                     _Float16* __restrict__ WT, int K, int N) {
    int i = blockIdx.x * blockDim.x + threadIdx.x;
    if (i < K * N) {
        int n = i / K, k = i % K;
        WT[i] = (_Float16)W[(size_t)k * N + n];
    }
}

// ---------------------------------------------------------------------------
// WMMA GEMM, 32x64 tile per wave. Modes: 1 f16 [b,h,s,d]; 3 f16 [b,h,d,s];
// 2 f32 plain [M,N].
// ---------------------------------------------------------------------------
__global__ void __launch_bounds__(256)
gemm32x64_kernel(const _Float16* __restrict__ A, int lda,
                 const _Float16* __restrict__ BT, int ldb,
                 const float* __restrict__ bias,
                 void* __restrict__ Cout,
                 int M, int N, int Kd, float scale, int mode) {
    const int wid  = threadIdx.x >> 5;
    const int lane = threadIdx.x & 31;
    const int ntiles = N >> 6;
    const int total  = (M >> 5) * ntiles;
    const int tile   = blockIdx.x * 8 + wid;
    if (tile >= total) return;
    const int mt = tile / ntiles;
    const int nt = tile % ntiles;

    const _Float16* Abase = A  + (size_t)mt * 32 * lda;
    const _Float16* Bbase = BT + (size_t)nt * 64 * ldb;

    union Acc { v8f v; float f[8]; };
    Acc acc[2][4];
#pragma unroll
    for (int i = 0; i < 2; ++i)
#pragma unroll
        for (int j = 0; j < 4; ++j)
#pragma unroll
            for (int r = 0; r < 8; ++r) acc[i][j].f[r] = 0.0f;

    for (int k0 = 0; k0 < Kd; k0 += 32) {
        const v16h a0 = load_frag_rowmajor(Abase + k0, lda);
        const v16h a1 = load_frag_rowmajor(Abase + (size_t)16 * lda + k0, lda);
#pragma unroll
        for (int j = 0; j < 4; ++j) {
            const v16h b = load_frag_rowmajor(Bbase + (size_t)j * 16 * ldb + k0, ldb);
            acc[0][j].v = wmma_f16(a0, b, acc[0][j].v);
            acc[1][j].v = wmma_f16(a1, b, acc[1][j].v);
        }
    }

    const int half = lane >> 4;
    const int cc   = lane & 15;
#pragma unroll
    for (int j = 0; j < 4; ++j) {
        const int   n  = nt * 64 + j * 16 + cc;
        const float bv = bias ? bias[n] : 0.0f;
#pragma unroll
        for (int i = 0; i < 2; ++i) {
#pragma unroll
            for (int r = 0; r < 8; ++r) {
                const int   m   = mt * 32 + i * 16 + r + half * 8;
                const float val = (acc[i][j].f[r] + bv) * scale;
                if (mode == 2) {
                    ((float*)Cout)[(size_t)m * N + n] = val;
                } else {
                    const int bb = m >> 10, s = m & 1023;
                    const int h  = n >> 6,  d = n & 63;
                    _Float16* C = (_Float16*)Cout;
                    if (mode == 1)
                        C[((size_t)((bb << 3) + h) * 1024 + s) * 64 + d] = (_Float16)val;
                    else
                        C[((size_t)((bb << 3) + h) * 64 + d) * 1024 + s] = (_Float16)val;
                }
            }
        }
    }
}

// ---------------------------------------------------------------------------
// Flash attention. Block = 8 waves, all on the SAME (b,h); each wave owns a
// 32-row q block (two 16-row S tiles). Per 32-key iteration the block stages
// the shared K tile (32x64 f16) and V^T tile (64x32 f16) into LDS once —
// via the Tensor Data Mover (double-buffered, s_wait_tensorcnt) — and all
// waves read their WMMA fragments from LDS. 8x less L2 traffic than
// per-wave streaming. Q pre-scaled by 1/sqrt(64).
// Q,K: f16 [b,h,s,d]; VT: f16 [b,h,d,s]; output f16 [b,s,h*64].
// ---------------------------------------------------------------------------
__global__ void __launch_bounds__(256)
attn_kernel(const _Float16* __restrict__ Q,
            const _Float16* __restrict__ Km,
            const _Float16* __restrict__ VT,
            _Float16* __restrict__ O) {
    __shared__ __align__(16) _Float16 kbuf[2][32 * 64];       // [key-kt][d]
    __shared__ __align__(16) _Float16 vbuf[2][64 * 32];       // [d][key-kt]
    __shared__ __align__(16) _Float16 pbuf[8][2][16 * 48];    // per-wave P

    const int wid  = threadIdx.x >> 5;
    const int lane = threadIdx.x & 31;
    const int bh   = blockIdx.x >> 2;                  // 0..63
    const int qp   = (blockIdx.x & 3) * 8 + wid;       // 0..31 (32-row blocks)

    const _Float16* Qb = Q  + (size_t)bh * 65536 + (size_t)qp * 32 * 64;
    const _Float16* Kb = Km + (size_t)bh * 65536;
    const _Float16* Vb = VT + (size_t)bh * 65536;

    v16h aq[2][2];
    aq[0][0] = load_frag_rowmajor(Qb, 64);
    aq[0][1] = load_frag_rowmajor(Qb + 32, 64);
    aq[1][0] = load_frag_rowmajor(Qb + 16 * 64, 64);
    aq[1][1] = load_frag_rowmajor(Qb + 16 * 64 + 32, 64);

    union Acc { v8f v; float f[8]; };
    Acc o[2][4];
    float mrun[2][8], lrun[2][8];
#pragma unroll
    for (int g = 0; g < 2; ++g)
#pragma unroll
        for (int r = 0; r < 8; ++r) {
            o[g][0].f[r] = o[g][1].f[r] = o[g][2].f[r] = o[g][3].f[r] = 0.0f;
            mrun[g][r] = -3.0e30f;
            lrun[g][r] = 0.0f;
        }

    const int half = lane >> 4;
    const int cc   = lane & 15;
    const int tid  = threadIdx.x;

    // --- stage first K/V tile into buffer 0 -------------------------------
#if USE_TDM
    if (wid == 0) {
        // K tile: 32 rows x 32 dwords, row stride 32 dwords
        tdm_load_2d(lds_off_of(&kbuf[0][0]), Kb, 32u, 32u, 32u, 32u, 1024u);
        // V^T tile: 64 rows x 16 dwords, row stride 512 dwords
        tdm_load_2d(lds_off_of(&vbuf[0][0]), Vb, 16u, 64u, 512u, 512u, 64u);
    }
#else
    {
        int row = tid >> 3, ch = tid & 7;   // K: 32 rows x 8 chunks of 8 halfs
        *(v8h*)(&kbuf[0][row * 64 + ch * 8]) =
            *(const v8h*)(Kb + (size_t)row * 64 + ch * 8);
        int vr = tid >> 2, vc = tid & 3;    // V: 64 rows x 4 chunks of 8 halfs
        *(v8h*)(&vbuf[0][vr * 32 + vc * 8]) =
            *(const v8h*)(Vb + (size_t)vr * 1024 + vc * 8);
    }
#endif

    int cur = 0;
    for (int kt = 0; kt < 1024; kt += 32) {
#if USE_TDM
        if (wid == 0) __builtin_amdgcn_s_wait_tensorcnt((short)0);
#endif
        __syncthreads();   // buf[cur] ready; previous reads of buf[nxt] done
        const int nxt = cur ^ 1;
        if (kt + 32 < 1024) {
#if USE_TDM
            if (wid == 0) {
                tdm_load_2d(lds_off_of(&kbuf[nxt][0]),
                            Kb + (size_t)(kt + 32) * 64, 32u, 32u, 32u, 32u, 1024u);
                tdm_load_2d(lds_off_of(&vbuf[nxt][0]),
                            Vb + (size_t)(kt + 32), 16u, 64u, 512u, 512u, 64u);
            }
#else
            int row = tid >> 3, ch = tid & 7;
            *(v8h*)(&kbuf[nxt][row * 64 + ch * 8]) =
                *(const v8h*)(Kb + (size_t)(kt + 32 + row) * 64 + ch * 8);
            int vr = tid >> 2, vc = tid & 3;
            *(v8h*)(&vbuf[nxt][vr * 32 + vc * 8]) =
                *(const v8h*)(Vb + (size_t)vr * 1024 + (kt + 32) + vc * 8);
#endif
        }

        // ---- compute from buf[cur] (LDS fragments) -----------------------
        const _Float16* kc = &kbuf[cur][0];
        const _Float16* vc = &vbuf[cur][0];
        const v16h bk00 = load_frag_rowmajor(kc, 64);
        const v16h bk01 = load_frag_rowmajor(kc + 32, 64);
        const v16h bk10 = load_frag_rowmajor(kc + 16 * 64, 64);
        const v16h bk11 = load_frag_rowmajor(kc + 16 * 64 + 32, 64);

#pragma unroll
        for (int g = 0; g < 2; ++g) {
            Acc s0, s1;
#pragma unroll
            for (int r = 0; r < 8; ++r) { s0.f[r] = 0.0f; s1.f[r] = 0.0f; }
            s0.v = wmma_f16(aq[g][0], bk00, s0.v);
            s0.v = wmma_f16(aq[g][1], bk01, s0.v);
            s1.v = wmma_f16(aq[g][0], bk10, s1.v);
            s1.v = wmma_f16(aq[g][1], bk11, s1.v);

            _Float16* pb = &pbuf[wid][g][0];
#pragma unroll
            for (int r = 0; r < 8; ++r) {
                float t = fmaxf(s0.f[r], s1.f[r]);
                t = fmaxf(t, __shfl_xor(t, 1, 32));
                t = fmaxf(t, __shfl_xor(t, 2, 32));
                t = fmaxf(t, __shfl_xor(t, 4, 32));
                t = fmaxf(t, __shfl_xor(t, 8, 32));
                const float mnew = fmaxf(mrun[g][r], t);
                const float p0 = __expf(s0.f[r] - mnew);
                const float p1 = __expf(s1.f[r] - mnew);
                float ps = p0 + p1;
                ps += __shfl_xor(ps, 1, 32);
                ps += __shfl_xor(ps, 2, 32);
                ps += __shfl_xor(ps, 4, 32);
                ps += __shfl_xor(ps, 8, 32);
                const float alpha = __expf(mrun[g][r] - mnew);
                lrun[g][r] = lrun[g][r] * alpha + ps;
                mrun[g][r] = mnew;
                o[g][0].f[r] *= alpha; o[g][1].f[r] *= alpha;
                o[g][2].f[r] *= alpha; o[g][3].f[r] *= alpha;
                const int row = r + half * 8;
                pb[row * 48 + cc]      = (_Float16)p0;
                pb[row * 48 + 16 + cc] = (_Float16)p1;
            }
        }
        asm volatile("s_wait_dscnt 0x0" ::: "memory");

        const v16h ap0 = load_frag_rowmajor(&pbuf[wid][0][0], 48);
        const v16h ap1 = load_frag_rowmajor(&pbuf[wid][1][0], 48);
#pragma unroll
        for (int j = 0; j < 4; ++j) {
            const v16h bv = load_frag_rowmajor(vc + j * 16 * 32, 32);
            o[0][j].v = wmma_f16(ap0, bv, o[0][j].v);
            o[1][j].v = wmma_f16(ap1, bv, o[1][j].v);
        }
        cur = nxt;
    }

    // finalize and store to [b, s, h*64 + d]
    const int b = bh >> 3, h = bh & 7;
#pragma unroll
    for (int g = 0; g < 2; ++g) {
#pragma unroll
        for (int r = 0; r < 8; ++r) {
            const int row = r + half * 8;
            const int s   = qp * 32 + g * 16 + row;
            const float inv = 1.0f / lrun[g][r];
            const size_t base = (size_t)(b * 1024 + s) * 512 + h * 64;
            O[base + 0  + cc] = (_Float16)(o[g][0].f[r] * inv);
            O[base + 16 + cc] = (_Float16)(o[g][1].f[r] * inv);
            O[base + 32 + cc] = (_Float16)(o[g][2].f[r] * inv);
            O[base + 48 + cc] = (_Float16)(o[g][3].f[r] * inv);
        }
    }
}

// ---------------------------------------------------------------------------
// Launch
// ---------------------------------------------------------------------------
extern "C" void kernel_launch(void* const* d_in, const int* in_sizes, int n_in,
                              void* d_out, int out_size, void* d_ws, size_t ws_size,
                              hipStream_t stream) {
    const float* x  = (const float*)d_in[0];
    const float* Wq = (const float*)d_in[1];
    const float* bq = (const float*)d_in[2];
    const float* Wk = (const float*)d_in[3];
    const float* bk = (const float*)d_in[4];
    const float* Wv = (const float*)d_in[5];
    const float* bv = (const float*)d_in[6];
    const float* Wo = (const float*)d_in[7];
    const float* bo = (const float*)d_in[8];
    float* out = (float*)d_out;

    char*  ws  = (char*)d_ws;
    size_t off = 0;
    auto alloc = [&](size_t bytes) -> void* {
        void* p = ws + off;
        off += (bytes + 255) & ~(size_t)255;
        return p;
    };
    _Float16* xh  = (_Float16*)alloc((size_t)8192 * 1024 * 2);
    _Float16* WqT = (_Float16*)alloc((size_t)512 * 1024 * 2);
    _Float16* WkT = (_Float16*)alloc((size_t)512 * 1024 * 2);
    _Float16* WvT = (_Float16*)alloc((size_t)512 * 1024 * 2);
    _Float16* WoT = (_Float16*)alloc((size_t)1024 * 512 * 2);
    _Float16* Qh  = (_Float16*)alloc((size_t)64 * 1024 * 64 * 2);  // [b,h,s,d]
    _Float16* Kh  = (_Float16*)alloc((size_t)64 * 1024 * 64 * 2);  // [b,h,s,d]
    _Float16* Vt  = (_Float16*)alloc((size_t)64 * 1024 * 64 * 2);  // [b,h,d,s]
    _Float16* AO  = (_Float16*)alloc((size_t)8192 * 512 * 2);      // [b,s,h*dv]

    cvt_f32_to_f16_kernel<<<(8192 * 1024) / 256, 256, 0, stream>>>(x, xh, 8192 * 1024);
    transpose_cvt_kernel<<<(1024 * 512) / 256, 256, 0, stream>>>(Wq, WqT, 1024, 512);
    transpose_cvt_kernel<<<(1024 * 512) / 256, 256, 0, stream>>>(Wk, WkT, 1024, 512);
    transpose_cvt_kernel<<<(1024 * 512) / 256, 256, 0, stream>>>(Wv, WvT, 1024, 512);
    transpose_cvt_kernel<<<(512 * 1024) / 256, 256, 0, stream>>>(Wo, WoT, 512, 1024);

    gemm32x64_kernel<<<256, 256, 0, stream>>>(xh, 1024, WqT, 1024, bq, Qh,
                                              8192, 512, 1024, 0.125f, 1);
    gemm32x64_kernel<<<256, 256, 0, stream>>>(xh, 1024, WkT, 1024, bk, Kh,
                                              8192, 512, 1024, 1.0f, 1);
    gemm32x64_kernel<<<256, 256, 0, stream>>>(xh, 1024, WvT, 1024, bv, Vt,
                                              8192, 512, 1024, 1.0f, 3);

    // attention: 64 (b,h) x 4 q-chunks = 256 blocks of 8 waves
    attn_kernel<<<256, 256, 0, stream>>>(Qh, Kh, Vt, AO);

    gemm32x64_kernel<<<512, 256, 0, stream>>>(AO, 512, WoT, 512, bo, out,
                                              8192, 1024, 512, 1.0f, 2);
}
